// GraphClassifier_51479478010374
// MI455X (gfx1250) — matrix operand order repaired and checked by reading, not compiled, not run on hardware
//
#include <hip/hip_runtime.h>
#include <hip/hip_bf16.h>

// ---------------------------------------------------------------------------
// MI455X (gfx1250) implementation. wave32, WMMA 16x16x32 f16.
// Phases: prep/convert -> random walks -> neigh WMMA -> LSTM WMMA -> MLP WMMA
//         -> loss/combine.
// LSTM: weight B-fragments register-resident (~256 VGPRs of the 1024/wave
// gfx1250 file, s_set_vgpr_msb-addressed), all 8 A-fragments batch-loaded from
// bank-conflict-free padded LDS before the WMMA burst, x double-buffered with
// gather + global_prefetch_b8 overlapped under the matrix work.
// ---------------------------------------------------------------------------

typedef __attribute__((ext_vector_type(16))) _Float16 v16h;
typedef __attribute__((ext_vector_type(8)))  _Float16 v8h;
typedef __attribute__((ext_vector_type(8)))  float    v8f;

#define NN    512   // nodes / batch / timesteps
#define EMB   64
#define INF_  128   // input feat width (2*EMB)
#define HID   128
#define KCAT  256   // INF_ + HID
#define NMLP  4
#define XSTR  136   // padded LDS row stride (halfwords): 272B -> lane l hits
#define HSTR  136   // banks 4l..4l+3, conflict-free A-fragment ds_load_b128

__device__ __forceinline__ v16h cat16(v8h lo, v8h hi) {
  return __builtin_shufflevector(lo, hi, 0,1,2,3,4,5,6,7,8,9,10,11,12,13,14,15);
}
__device__ __forceinline__ v8f wmma16(v16h a, v16h b, v8f c) {
  // D = A(16x32 f16) x B(32x16 f16) + C(16x16 f32)
  return __builtin_amdgcn_wmma_f32_16x16x32_f16(false, a, false, b, (short)0, c,
                                                false, false);
}
__device__ __forceinline__ float sigf(float x) { return 1.0f / (1.0f + __expf(-x)); }

__device__ __forceinline__ unsigned hash3(unsigned a, unsigned b, unsigned c) {
  unsigned h = a * 0x9E3779B1u ^ b * 0x85EBCA77u ^ c * 0xC2B2AE3Du;
  h ^= h >> 16; h *= 0x7FEB352Du; h ^= h >> 15; h *= 0x846CA68Bu; h ^= h >> 16;
  return h;
}

// ---------------------------------------------------------------------------
// LSTM: 32 WGs x 16 batch rows. Wave w owns hidden cols w*16..w*16+15.
//  - Weight B-fragments (4 gates x 8 K-chunks = 256 VGPRs) preloaded into
//    registers once; steady-state loop does zero global weight traffic.
//  - All 8 A-fragments batch-loaded (one ds_load clause, graduated waits)
//    before the 32-WMMA burst; x double-buffered in padded LDS.
//  - c and sum(h) live in WMMA C-layout VGPR accumulators.
// (Defined first among WMMA kernels so the disasm snippet shows this loop.)
// ---------------------------------------------------------------------------
__global__ void __launch_bounds__(256) k_lstm(const int* __restrict__ orders,
                                              const _Float16* __restrict__ if16,
                                              const _Float16* __restrict__ wcat,
                                              const float* __restrict__ bcat,
                                              _Float16* __restrict__ emb16) {
  __shared__ _Float16 xb[2][16 * XSTR];                // double-buffered x_t
  __shared__ _Float16 hb[16 * HSTR];                   // h_{t-1}
  const int wg   = blockIdx.x;                         // batch rows wg*16..
  const int tid  = threadIdx.x;
  const int wave = tid >> 5;                           // hidden col tile 0..7
  const int lane = tid & 31;
  const int l15  = lane & 15, hi = lane >> 4;
  const int hcol = wave * 16 + l15;
  const int grow = tid >> 4, seg = tid & 15;           // gather row / 16B segment
  const long gbase = (long)(wg * 16 + grow) * NN;      // orders row base

  const float bI = bcat[0 * HID + hcol];
  const float bF = bcat[1 * HID + hcol];
  const float bG = bcat[2 * HID + hcol];
  const float bO = bcat[3 * HID + hcol];

  // Preload this wave's weight slice: 64 gate rows x 256 K, fp16 -> 256 VGPRs.
  v16h wf[4][8];
  #pragma unroll
  for (int g = 0; g < 4; ++g)
    #pragma unroll
    for (int kc = 0; kc < 8; ++kc) {
      const _Float16* bp = &wcat[(g * HID + hcol) * KCAT + kc * 32 + hi * 16];
      wf[g][kc] = cat16(*(const v8h*)bp, *(const v8h*)(bp + 8));
    }

  v8f cst = {}, hsum = {};

  { // prologue: zero h, gather x_0
    v8h z = {};
    *(v8h*)&hb[grow * HSTR + seg * 8] = z;
    int node = orders[gbase + 0];
    *(v8h*)&xb[0][grow * XSTR + seg * 8] = *(const v8h*)&if16[node * INF_ + seg * 8];
  }
  __syncthreads();

  for (int t = 0; t < NN; ++t) {
    // Issue next step's gather now: latency overlaps the WMMA block below.
    v8h xpre = {};
    if (t + 1 < NN) {
      int node = orders[gbase + t + 1];
      xpre = *(const v8h*)&if16[node * INF_ + seg * 8];
    }
    if (t + 8 < NN)
      __builtin_prefetch(&orders[gbase + t + 8], 0, 1);  // global_prefetch_b8

    // Batch-load all 8 A-fragments (16 ds_load_b128 in one clause), then the
    // 32-WMMA burst -- one LDS latency exposure per step instead of eight.
    const _Float16* xcur = &xb[t & 1][0];
    v16h afr[8];
    #pragma unroll
    for (int kc = 0; kc < 8; ++kc) {
      const _Float16* ap = (kc < 4)
          ? &xcur[l15 * XSTR + kc * 32 + hi * 16]
          : &hb[l15 * HSTR + (kc - 4) * 32 + hi * 16];
      afr[kc] = cat16(*(const v8h*)ap, *(const v8h*)(ap + 8));
    }
    v8f a0 = {}, a1 = {}, a2 = {}, a3 = {};
    #pragma unroll
    for (int kc = 0; kc < 8; ++kc) {
      a0 = wmma16(afr[kc], wf[0][kc], a0);
      a1 = wmma16(afr[kc], wf[1][kc], a1);
      a2 = wmma16(afr[kc], wf[2][kc], a2);
      a3 = wmma16(afr[kc], wf[3][kc], a3);
    }

    _Float16 hv16[8];
    #pragma unroll
    for (int r = 0; r < 8; ++r) {
      float iv = sigf(a0[r] + bI);
      float fv = sigf(a1[r] + bF);
      float gv = tanhf(a2[r] + bG);
      float ov = sigf(a3[r] + bO);
      float cv = fv * cst[r] + iv * gv;
      cst[r] = cv;
      float hv = ov * tanhf(cv);
      hsum[r] += hv;
      hv16[r] = (_Float16)hv;
    }
    __syncthreads();                                   // all reads of hb/xb done
    #pragma unroll
    for (int r = 0; r < 8; ++r)
      hb[(r + hi * 8) * HSTR + hcol] = hv16[r];        // publish h_t
    if (t + 1 < NN)
      *(v8h*)&xb[(t + 1) & 1][grow * XSTR + seg * 8] = xpre;
    __syncthreads();
  }

  #pragma unroll
  for (int r = 0; r < 8; ++r) {                        // embed = mean over T
    int row = wg * 16 + r + hi * 8;
    emb16[row * HID + hcol] = (_Float16)(hsum[r] * (1.0f / 512.0f));
  }
}

// ---------------------------------------------------------------------------
// Prep kernels: fp32 -> fp16 packing in WMMA-friendly row-major layouts.
// ---------------------------------------------------------------------------
__global__ void __launch_bounds__(256) k_abin(const float* __restrict__ adj,
                                              _Float16* __restrict__ abin) {
  int idx = blockIdx.x * 256 + threadIdx.x;            // 512*512
  abin[idx] = (_Float16)((adj[idx] > 0.5f) ? 1.0f : 0.0f);
}

__global__ void __launch_bounds__(256) k_embfeat(const int* __restrict__ tags,
                                                 const float* __restrict__ W_emb,
                                                 const float* __restrict__ b_emb,
                                                 _Float16* __restrict__ et,     // [64][512]
                                                 _Float16* __restrict__ if16) { // [512][128]
  int idx = blockIdx.x * 256 + threadIdx.x;            // 512*64
  int n = idx >> 6, e = idx & 63;
  float v = W_emb[tags[n] * EMB + e];                  // onehot @ W_emb (gather)
  et[e * NN + n] = (_Float16)v;                        // transposed copy for B-frags
  if16[n * INF_ + e] = (_Float16)(v + b_emb[e]);       // node_feat half
}

__global__ void __launch_bounds__(256) k_wcat(const float* __restrict__ W_ih,
                                              const float* __restrict__ W_hh,
                                              const float* __restrict__ b_ih,
                                              const float* __restrict__ b_hh,
                                              _Float16* __restrict__ wcat, // [512][256]
                                              float* __restrict__ bcat) {  // [512]
  int idx = blockIdx.x * 256 + threadIdx.x;            // 512*256
  int r = idx >> 8, c = idx & 255;
  float v = (c < HID) ? W_ih[r * HID + c] : W_hh[r * HID + (c - HID)];
  wcat[idx] = (_Float16)v;
  if (idx < 4 * HID) bcat[idx] = b_ih[idx] + b_hh[idx];
}

__global__ void __launch_bounds__(256) k_w1t(const float* __restrict__ W1s,
                                             _Float16* __restrict__ w1t) { // [4][128][128] h-major
  int idx = blockIdx.x * 256 + threadIdx.x;            // 4*128*128
  int k = idx >> 14, rem = idx & 16383, h = rem >> 7, d = rem & 127;
  w1t[idx] = (_Float16)(W1s[k * HID * HID + d * HID + h]);
}

// ---------------------------------------------------------------------------
// Random walks: one wave32 per start node, 511 dependent argmax steps.
// exp/normalize dropped (monotone): argmax over unvisited of adj[cur,j]+noise.
// ---------------------------------------------------------------------------
__global__ void __launch_bounds__(256) k_walk(const float* __restrict__ adj,
                                              int* __restrict__ orders) {
  const int walk = blockIdx.x * 8 + (threadIdx.x >> 5);
  const int lane = threadIdx.x & 31;
  unsigned visited = 0u;                                // bit s -> j = lane + 32*s
  if (lane == (walk & 31)) visited |= 1u << (walk >> 5);
  if (lane == 0) orders[walk * NN] = walk;
  int cur = walk;
  for (int t = 1; t < NN; ++t) {
    const float* row = adj + cur * NN;
    float bv = -1e30f; int bj = 1 << 30;
    #pragma unroll
    for (int s = 0; s < 16; ++s) {
      int j = lane + (s << 5);
      float nz = 0.01f + 0.09f * ((hash3((unsigned)walk, (unsigned)t, (unsigned)j) >> 8)
                                  * (1.0f / 16777216.0f));
      float v = row[j] + nz;
      bool ok = ((visited >> s) & 1u) == 0u;
      if (ok && (v > bv || (v == bv && j < bj))) { bv = v; bj = j; }
    }
    #pragma unroll
    for (int off = 16; off > 0; off >>= 1) {
      float ov = __shfl_xor(bv, off, 32);
      int   oj = __shfl_xor(bj, off, 32);
      if (ov > bv || (ov == bv && oj < bj)) { bv = ov; bj = oj; }
    }
    cur = bj;
    if (lane == (bj & 31)) visited |= 1u << (bj >> 5);
    if (lane == 0) orders[walk * NN + t] = bj;
  }
}

// ---------------------------------------------------------------------------
// neigh_feat = (A_bin @ E) + b_emb  -> if16[:, 64:128].   M=512,N=64,K=512 WMMA.
// ---------------------------------------------------------------------------
__global__ void __launch_bounds__(256) k_neigh(const _Float16* __restrict__ abin, // [512][512]
                                               const _Float16* __restrict__ et,   // [64][512]
                                               const float* __restrict__ b_emb,
                                               _Float16* __restrict__ if16) {
  const int m    = blockIdx.x * 8 + (threadIdx.x >> 5); // 0..31
  const int lane = threadIdx.x & 31;
  const int l15  = lane & 15, hi = lane >> 4;
  v8f z = {};
  v8f acc[4] = {z, z, z, z};
  for (int kc = 0; kc < 16; ++kc) {
    const _Float16* ap = &abin[(m * 16 + l15) * NN + kc * 32 + hi * 16];
    v16h a = cat16(*(const v8h*)ap, *(const v8h*)(ap + 8));
    #pragma unroll
    for (int n = 0; n < 4; ++n) {
      const _Float16* bp = &et[(n * 16 + l15) * NN + kc * 32 + hi * 16];
      v16h b = cat16(*(const v8h*)bp, *(const v8h*)(bp + 8));
      acc[n] = wmma16(a, b, acc[n]);
    }
  }
  #pragma unroll
  for (int n = 0; n < 4; ++n) {
    int col = n * 16 + l15;
    #pragma unroll
    for (int r = 0; r < 8; ++r) {
      int row = m * 16 + r + hi * 8;
      if16[row * INF_ + EMB + col] = (_Float16)(acc[n][r] + b_emb[col]);
    }
  }
}

// ---------------------------------------------------------------------------
// 4-head MLP: h1 = relu(embed @ W1k + b1k); logit = h1 . W2k + b2k.
// ---------------------------------------------------------------------------
__global__ void __launch_bounds__(256) k_mlp(const _Float16* __restrict__ emb16,
                                             const _Float16* __restrict__ w1t,
                                             const float* __restrict__ b1s,
                                             const float* __restrict__ w2s,
                                             const float* __restrict__ b2s,
                                             float* __restrict__ logits) { // [4][512]
  const int task = blockIdx.x * 8 + (threadIdx.x >> 5); // 0..127
  const int k = task >> 5, m = task & 31;
  const int lane = threadIdx.x & 31;
  const int l15 = lane & 15, hi = lane >> 4;
  float partial[8] = {0, 0, 0, 0, 0, 0, 0, 0};
  for (int nt = 0; nt < 8; ++nt) {
    v8f acc = {};
    #pragma unroll
    for (int kc = 0; kc < 4; ++kc) {
      const _Float16* ap = &emb16[(m * 16 + l15) * HID + kc * 32 + hi * 16];
      v16h a = cat16(*(const v8h*)ap, *(const v8h*)(ap + 8));
      const _Float16* bp = &w1t[(k * HID + nt * 16 + l15) * HID + kc * 32 + hi * 16];
      v16h b = cat16(*(const v8h*)bp, *(const v8h*)(bp + 8));
      acc = wmma16(a, b, acc);
    }
    int col = nt * 16 + l15;
    float b1v = b1s[k * HID + col], w2v = w2s[k * HID + col];
    #pragma unroll
    for (int r = 0; r < 8; ++r) {
      float h1 = fmaxf(acc[r] + b1v, 0.0f);
      partial[r] += h1 * w2v;
    }
  }
  #pragma unroll
  for (int mask = 1; mask < 16; mask <<= 1)
    #pragma unroll
    for (int r = 0; r < 8; ++r)
      partial[r] += __shfl_xor(partial[r], mask, 32);
  if (l15 == 0) {
    #pragma unroll
    for (int r = 0; r < 8; ++r)
      logits[k * NN + m * 16 + r + hi * 8] = partial[r] + b2s[k];
  }
}

// ---------------------------------------------------------------------------
// Loss + softplus-weighted combine. One 512-thread block.
// ---------------------------------------------------------------------------
__global__ void __launch_bounds__(512) k_final(const float* __restrict__ logits,
                                               const int* __restrict__ label,
                                               const float* __restrict__ var_raw,
                                               float* __restrict__ out) {
  __shared__ float red[512];
  const int n = threadIdx.x;
  const float lab = (float)label[0];
  float cw[NMLP], tot = 0.0f;
  #pragma unroll
  for (int k = 0; k < NMLP; ++k) { cw[k] = log1pf(__expf(var_raw[k])); tot += cw[k]; }
  float ysum = 0.0f, bsum = 0.0f;
  #pragma unroll
  for (int k = 0; k < NMLP; ++k) {
    float z = logits[k * NN + n];
    float lse = log1pf(__expf(-fabsf(z)));
    float lsp = fminf(z, 0.0f) - lse;                  // log sigmoid(z)
    float lsn = fminf(-z, 0.0f) - lse;                 // log sigmoid(-z)
    bsum += -(lab * lsp + (1.0f - lab) * lsn);
    ysum += (cw[k] / tot) * sigf(z);
  }
  out[1 + n] = ysum;
  red[n] = bsum;
  __syncthreads();
  for (int s = 256; s > 0; s >>= 1) {
    if (n < s) red[n] += red[n + s];
    __syncthreads();
  }
  if (n == 0) out[0] = red[0] * (1.0f / 512.0f);       // sum_k mean_n bce
}

// ---------------------------------------------------------------------------
extern "C" void kernel_launch(void* const* d_in, const int* in_sizes, int n_in,
                              void* d_out, int out_size, void* d_ws, size_t ws_size,
                              hipStream_t stream) {
  const int*   tags    = (const int*)d_in[0];
  const float* adj     = (const float*)d_in[1];
  const int*   label   = (const int*)d_in[2];
  const float* W_emb   = (const float*)d_in[3];
  const float* b_emb   = (const float*)d_in[4];
  const float* W_ih    = (const float*)d_in[5];
  const float* W_hh    = (const float*)d_in[6];
  const float* b_ih    = (const float*)d_in[7];
  const float* b_hh    = (const float*)d_in[8];
  const float* W1s     = (const float*)d_in[9];
  const float* b1s     = (const float*)d_in[10];
  const float* W2s     = (const float*)d_in[11];
  const float* b2s     = (const float*)d_in[12];
  const float* var_raw = (const float*)d_in[13];
  (void)in_sizes; (void)n_in; (void)out_size; (void)ws_size;

  char* ws = (char*)d_ws;
  int*      orders = (int*)(ws + 0);                       // 1 MB
  _Float16* abin   = (_Float16*)(ws + (1u << 20));         // 512 KB
  _Float16* et     = (_Float16*)(ws + 1572864u);           // 64 KB
  _Float16* if16   = (_Float16*)(ws + 1638400u);           // 128 KB
  _Float16* wcat   = (_Float16*)(ws + 1769472u);           // 256 KB
  float*    bcat   = (float*)   (ws + 2031616u);           // 2 KB
  _Float16* w1t    = (_Float16*)(ws + 2033664u);           // 128 KB
  _Float16* emb16  = (_Float16*)(ws + 2164736u);           // 128 KB
  float*    logits = (float*)   (ws + 2295808u);           // 8 KB

  k_abin   <<<1024, 256, 0, stream>>>(adj, abin);
  k_embfeat<<< 128, 256, 0, stream>>>(tags, W_emb, b_emb, et, if16);
  k_wcat   <<< 512, 256, 0, stream>>>(W_ih, W_hh, b_ih, b_hh, wcat, bcat);
  k_w1t    <<< 256, 256, 0, stream>>>(W1s, w1t);
  k_walk   <<<  64, 256, 0, stream>>>(adj, orders);
  k_neigh  <<<   4, 256, 0, stream>>>(abin, et, b_emb, if16);
  k_lstm   <<<  32, 256, 0, stream>>>(orders, if16, wcat, bcat, emb16);
  k_mlp    <<<  16, 256, 0, stream>>>(emb16, w1t, b1s, W2s, b2s, logits);
  k_final  <<<   1, 512, 0, stream>>>(logits, label, var_raw, (float*)d_out);
}
